// TemporalBlock_58093727645828
// MI455X (gfx1250) — matrix operand order repaired and checked by reading, not compiled, not run on hardware
//
#include <hip/hip_runtime.h>
#include <math.h>

typedef _Float16 half_t;
typedef __attribute__((ext_vector_type(16))) _Float16 v16h;
typedef __attribute__((ext_vector_type(8)))  _Float16 v8h;
typedef __attribute__((ext_vector_type(8)))  float    v8f;

#define HH 32
#define WW 32
#define HWN 1024
#define DIM 256
#define NH 8
#define DH 32
#define BB 4
#define TQ 9
#define TT 8
#define MLPD 1024
#define NEGV (-1e9f)
#define SCALE 0.17677669529663687f  // 1/sqrt(32)

// ---------------------------------------------------------------------------
// Weight transpose + f16 convert: W (K x N, fp32, row-major) -> Wt (N x K, f16)
// ---------------------------------------------------------------------------
__global__ void wtrans_kernel(const float* __restrict__ W, half_t* __restrict__ Wt,
                              int K, int N) {
  int i = blockIdx.x * blockDim.x + threadIdx.x;
  if (i >= K * N) return;
  int k = i / N, n = i % N;
  Wt[(size_t)n * K + k] = (half_t)W[i];
}

// ---------------------------------------------------------------------------
// LayerNorm over last dim (256). One row per 256-thread block.
// remap: 0 = identity row, 1 = spatial (row over B*T*HW, source has T=9 frames),
//        2 = temporal ((b,p,t) row order reading (b,t,p) layout)
// ---------------------------------------------------------------------------
__global__ void ln_kernel(const float* __restrict__ in, const float* __restrict__ g,
                          const float* __restrict__ bta, half_t* __restrict__ out,
                          int remap) {
  int row = blockIdx.x;
  int d = threadIdx.x;
  size_t irow = row;
  if (remap == 1) {
    irow = (size_t)row + (size_t)((row >> 13) << 10);          // + b*1024
  } else if (remap == 2) {
    int b = row / 9216, rem = row % 9216, p = rem / 9, t = rem % 9;
    irow = ((size_t)b * TQ + t) * HWN + p;
  }
  float x = in[irow * DIM + d];
  float s = x, q = x * x;
  for (int off = 16; off; off >>= 1) {
    s += __shfl_xor(s, off, 32);
    q += __shfl_xor(q, off, 32);
  }
  __shared__ float ss[8], sq[8];
  int wv = threadIdx.x >> 5, ln = threadIdx.x & 31;
  if (ln == 0) { ss[wv] = s; sq[wv] = q; }
  __syncthreads();
  if (threadIdx.x == 0) {
    float ts = 0.f, tq = 0.f;
    for (int i = 0; i < 8; i++) { ts += ss[i]; tq += sq[i]; }
    ss[0] = ts; sq[0] = tq;
  }
  __syncthreads();
  float mean = ss[0] * (1.f / DIM);
  float var  = sq[0] * (1.f / DIM) - mean * mean;
  float y = (x - mean) * rsqrtf(var + 1e-5f) * g[d] + bta[d];
  out[(size_t)row * DIM + d] = (half_t)y;
}

// ---------------------------------------------------------------------------
// WMMA GEMM: C[M,N] = A[M,K] (f16 row-major) x Wt[N,K]^T (f16) + bias
// One wave32 per 16x16 output tile; K loop of v_wmma_f32_16x16x32_f16.
// A fragment (16-bit A 16x32): lane holds row M=lane&15,
//   K = 16 contiguous values starting at klo = (lane>=16) ? 16 : 0.
// B fragment (dense 16-bit B 32x16): lane holds col N=lane&15,
//   K = 16 contiguous values starting at (lane>=16 ? 16 : 0).
// D layout: lane<16 -> N=lane, M=e; lane>=16 -> N=lane-16, M=8+e.
// act: 0 = none, 1 = exact GELU. remap (rows of out_f/res_f): as ln_kernel.
// ---------------------------------------------------------------------------
__global__ void gemm_wmma(const half_t* __restrict__ A, const half_t* __restrict__ Wt,
                          const float* __restrict__ bias,
                          half_t* __restrict__ out_h,
                          float* __restrict__ out_f, const float* __restrict__ res_f,
                          int M, int N, int K, int remap, int act) {
  int wave = (blockIdx.x << 3) + (threadIdx.x >> 5);
  int ntiles = N >> 4;
  int tM = wave / ntiles;
  int tN = wave % ntiles;
  if (tM * 16 >= M) return;

  int lane = threadIdx.x & 31;
  int l15 = lane & 15;
  int hi = (lane & 16) ? 1 : 0;

  const size_t aRow = (size_t)(tM * 16 + l15) * K;
  const size_t bRow = (size_t)(tN * 16 + l15) * K;
  const int aklo = hi ? 16 : 0;
  const int bklo = hi ? 16 : 0;

  v8f c = {};
  for (int kb = 0; kb < K; kb += 32) {
    const half_t* ap = A + aRow + kb + aklo;
    v8h a0 = *(const v8h*)(ap);
    v8h a1 = *(const v8h*)(ap + 8);
    v16h av;
#pragma unroll
    for (int i = 0; i < 8; i++) { av[i] = a0[i]; av[i + 8] = a1[i]; }
    v16h bv = *(const v16h*)(Wt + bRow + kb + bklo);
    c = __builtin_amdgcn_wmma_f32_16x16x32_f16(false, av, false, bv,
                                               (short)0, c, false, false);
  }

  int ncol = tN * 16 + l15;
  float bb = bias[ncol];
  int mbase = tM * 16 + (hi ? 8 : 0);
#pragma unroll
  for (int e = 0; e < 8; e++) {
    int m = mbase + e;
    float val = c[e] + bb;
    if (act == 1) val = 0.5f * val * (1.f + erff(val * 0.70710678118654752f));
    if (out_h) out_h[(size_t)m * N + ncol] = (half_t)val;
    if (out_f) {
      size_t orow = m;
      if (remap == 1) {
        orow = (size_t)m + (size_t)((m >> 13) << 10);
      } else if (remap == 2) {
        int b = m / 9216, rem = m % 9216, p = rem / 9, t = rem % 9;
        orow = ((size_t)b * TQ + t) * HWN + p;
      }
      size_t oi = orow * (size_t)N + ncol;
      float r = res_f ? res_f[oi] : 0.f;
      out_f[oi] = val + r;
    }
  }
}

// ---------------------------------------------------------------------------
// Spatial 3x3 windowed attention. One wave per (bt, pixel, head); lane = dim.
// ---------------------------------------------------------------------------
__global__ void spatial_attn(const half_t* __restrict__ Q, const half_t* __restrict__ Kk,
                             const half_t* __restrict__ V, half_t* __restrict__ O) {
  int wid = (blockIdx.x << 3) + (threadIdx.x >> 5);
  int lane = threadIdx.x & 31;
  int head = wid & 7;
  int p = (wid >> 3) & (HWN - 1);
  int bt = wid >> 13;
  if (bt >= BB * TT) return;

  size_t base = ((size_t)(bt * HWN + p)) * DIM + head * DH + lane;
  float qv = (float)Q[base];
  int y = p >> 5, x = p & 31;

  float sc[9];
  int np[9];
  float mx = -1e30f;
  for (int i = 0; i < 3; i++)
    for (int j = 0; j < 3; j++) {
      int idx = i * 3 + j;
      int ny = y + i - 1, nx = x + j - 1;
      if ((unsigned)ny > 31u || (unsigned)nx > 31u) { sc[idx] = NEGV; np[idx] = -1; continue; }
      int pp = ny * WW + nx;
      np[idx] = pp;
      float kv = (float)Kk[((size_t)(bt * HWN + pp)) * DIM + head * DH + lane];
      float s = qv * kv;
      for (int off = 16; off; off >>= 1) s += __shfl_xor(s, off, 32);
      s *= SCALE;
      sc[idx] = s;
      if (s > mx) mx = s;
    }
  float den = 0.f, w[9];
  for (int idx = 0; idx < 9; idx++) {
    float e = (np[idx] < 0) ? 0.f : expf(sc[idx] - mx);
    w[idx] = e; den += e;
  }
  float inv = 1.f / den, acc = 0.f;
  for (int idx = 0; idx < 9; idx++) {
    if (np[idx] < 0) continue;
    acc += w[idx] * inv * (float)V[((size_t)(bt * HWN + np[idx])) * DIM + head * DH + lane];
  }
  O[base] = (half_t)acc;
}

// ---------------------------------------------------------------------------
// Copy frame t=8 of query into X (fp32 identity copy of those positions).
// ---------------------------------------------------------------------------
__global__ void copy_last_frame(const float* __restrict__ query, float* __restrict__ X) {
  int i = blockIdx.x * blockDim.x + threadIdx.x;
  if (i >= BB * HWN * DIM) return;
  int b = i / (HWN * DIM);
  int r = i % (HWN * DIM);
  size_t gi = ((size_t)(b * TQ + TT)) * HWN * DIM + r;
  X[gi] = query[gi];
}

// ---------------------------------------------------------------------------
// RoPE (split-half) applied in-place to temporal Q and K ((b,p,t) rows).
// ---------------------------------------------------------------------------
__global__ void rope_kernel(half_t* __restrict__ Q, half_t* __restrict__ Kk) {
  int i = blockIdx.x * blockDim.x + threadIdx.x;
  if (i >= BB * HWN * TQ * NH * 16) return;
  int d = i & 15;
  int head = (i >> 4) & 7;
  int row = i >> 7;
  int t = row % TQ;
  float inv = powf(10000.f, -(2.f * (float)d) / 32.f);
  float ang = (float)t * inv;
  float cc = cosf(ang), ssn = sinf(ang);
  size_t i1 = (size_t)row * DIM + head * DH + d;
  size_t i2 = i1 + 16;
  float q1 = (float)Q[i1], q2 = (float)Q[i2];
  Q[i1] = (half_t)(q1 * cc - q2 * ssn);
  Q[i2] = (half_t)(q1 * ssn + q2 * cc);
  float k1 = (float)Kk[i1], k2 = (float)Kk[i2];
  Kk[i1] = (half_t)(k1 * cc - k2 * ssn);
  Kk[i2] = (half_t)(k1 * ssn + k2 * cc);
}

// ---------------------------------------------------------------------------
// Temporal masked attention (T=9). One wave per (b, pixel, head); lane = dim.
// ---------------------------------------------------------------------------
__global__ void temporal_attn(const half_t* __restrict__ Q, const half_t* __restrict__ Kk,
                              const half_t* __restrict__ V,
                              const unsigned char* __restrict__ mask,
                              half_t* __restrict__ O) {
  int wid = (blockIdx.x << 3) + (threadIdx.x >> 5);
  int lane = threadIdx.x & 31;
  int head = wid & 7;
  int p = (wid >> 3) & (HWN - 1);
  int b = wid >> 13;
  if (b >= BB) return;

  size_t rb = ((size_t)(b * HWN + p)) * TQ;
  for (int tq = 0; tq < TQ; tq++) {
    float qv = (float)Q[(rb + tq) * DIM + head * DH + lane];
    float sc[TQ];
    float mx = -1e30f;
    for (int tk = 0; tk < TQ; tk++) {
      float kv = (float)Kk[(rb + tk) * DIM + head * DH + lane];
      float s = qv * kv;
      for (int off = 16; off; off >>= 1) s += __shfl_xor(s, off, 32);
      s *= SCALE;
      if (mask[tq * TQ + tk]) s = NEGV;
      sc[tk] = s;
      if (s > mx) mx = s;
    }
    float den = 0.f, w[TQ];
    for (int tk = 0; tk < TQ; tk++) { float e = expf(sc[tk] - mx); w[tk] = e; den += e; }
    float inv = 1.f / den, acc = 0.f;
    for (int tk = 0; tk < TQ; tk++)
      acc += w[tk] * inv * (float)V[(rb + tk) * DIM + head * DH + lane];
    O[(rb + tq) * DIM + head * DH + lane] = (half_t)acc;
  }
}

// ---------------------------------------------------------------------------
// CLS frame averaging over spatial positions, then broadcast.
// ---------------------------------------------------------------------------
__global__ void cls_avg(const float* __restrict__ X2, float* __restrict__ avg) {
  int b = blockIdx.x;            // 4 blocks
  int d = threadIdx.x;           // 256 threads
  float s = 0.f;
  for (int p = 0; p < HWN; p++)
    s += X2[((size_t)(b * TQ) * HWN + p) * DIM + d];
  avg[b * DIM + d] = s * (1.f / HWN);
}

__global__ void cls_bcast(float* __restrict__ X2, const float* __restrict__ avg) {
  int i = blockIdx.x * blockDim.x + threadIdx.x;
  if (i >= BB * HWN * DIM) return;
  int d = i & 255;
  int p = (i >> 8) & (HWN - 1);
  int b = i >> 18;
  X2[((size_t)(b * TQ) * HWN + p) * DIM + d] = avg[b * DIM + d];
}

// ===========================================================================
// Host launch
// ===========================================================================
extern "C" void kernel_launch(void* const* d_in, const int* in_sizes, int n_in,
                              void* d_out, int out_size, void* d_ws, size_t ws_size,
                              hipStream_t stream) {
  (void)in_sizes; (void)n_in; (void)out_size; (void)ws_size;

  const float* query = (const float*)d_in[0];
  const unsigned char* tmask = (const unsigned char*)d_in[2];
  const float* sln_g = (const float*)d_in[3];
  const float* sln_b = (const float*)d_in[4];
  const float* s_wq = (const float*)d_in[5];  const float* s_bq = (const float*)d_in[6];
  const float* s_wk = (const float*)d_in[7];  const float* s_bk = (const float*)d_in[8];
  const float* s_wv = (const float*)d_in[9];  const float* s_bv = (const float*)d_in[10];
  const float* s_wo = (const float*)d_in[11]; const float* s_bo = (const float*)d_in[12];
  const float* tln_g = (const float*)d_in[13];
  const float* tln_b = (const float*)d_in[14];
  const float* t_wq = (const float*)d_in[15]; const float* t_bq = (const float*)d_in[16];
  const float* t_wk = (const float*)d_in[17]; const float* t_bk = (const float*)d_in[18];
  const float* t_wv = (const float*)d_in[19]; const float* t_bv = (const float*)d_in[20];
  const float* t_wo = (const float*)d_in[21]; const float* t_bo = (const float*)d_in[22];
  const float* mln_g = (const float*)d_in[23];
  const float* mln_b = (const float*)d_in[24];
  const float* w1 = (const float*)d_in[25];   const float* b1 = (const float*)d_in[26];
  const float* w2 = (const float*)d_in[27];   const float* b2 = (const float*)d_in[28];

  float* out = (float*)d_out;   // doubles as X2 (post-temporal activation)
  char* ws = (char*)d_ws;

  // workspace layout (bytes)
  const size_t X_BYTES   = (size_t)BB * TQ * HWN * DIM * 4;      // 37,748,736
  const size_t LN_BYTES  = (size_t)BB * TQ * HWN * DIM * 2;      // 18,874,368 (f16, 36864 rows)
  const size_t QB        = LN_BYTES;                             // per q/k/v/o f16 buffer
  const size_t X_OFF     = 0;
  const size_t LN_OFF    = X_OFF + X_BYTES;
  const size_t QKVO_OFF  = LN_OFF + LN_BYTES;                    // H (MLP hidden) aliases all 4
  const size_t AVG_OFF   = QKVO_OFF + 4 * QB;
  const size_t WT_OFF    = AVG_OFF + 4096;

  float*  X    = (float*)(ws + X_OFF);
  half_t* LN   = (half_t*)(ws + LN_OFF);
  half_t* Qb   = (half_t*)(ws + QKVO_OFF);
  half_t* Kb   = (half_t*)(ws + QKVO_OFF + QB);
  half_t* Vb   = (half_t*)(ws + QKVO_OFF + 2 * QB);
  half_t* Ob   = (half_t*)(ws + QKVO_OFF + 3 * QB);
  half_t* Hb   = (half_t*)(ws + QKVO_OFF);                       // 36864x1024 f16 = 4*QB
  float*  AVG  = (float*)(ws + AVG_OFF);

  half_t* s_wqT = (half_t*)(ws + WT_OFF);
  half_t* s_wkT = s_wqT + DIM * DIM;
  half_t* s_wvT = s_wkT + DIM * DIM;
  half_t* s_woT = s_wvT + DIM * DIM;
  half_t* t_wqT = s_woT + DIM * DIM;
  half_t* t_wkT = t_wqT + DIM * DIM;
  half_t* t_wvT = t_wkT + DIM * DIM;
  half_t* t_woT = t_wvT + DIM * DIM;
  half_t* w1T   = t_woT + DIM * DIM;          // 1024 x 256
  half_t* w2T   = w1T + DIM * MLPD;           // 256 x 1024

  const int TPB = 256;

  // --- weight conversion/transposition ---
  {
    int n = DIM * DIM, g = (n + TPB - 1) / TPB;
    wtrans_kernel<<<g, TPB, 0, stream>>>(s_wq, s_wqT, DIM, DIM);
    wtrans_kernel<<<g, TPB, 0, stream>>>(s_wk, s_wkT, DIM, DIM);
    wtrans_kernel<<<g, TPB, 0, stream>>>(s_wv, s_wvT, DIM, DIM);
    wtrans_kernel<<<g, TPB, 0, stream>>>(s_wo, s_woT, DIM, DIM);
    wtrans_kernel<<<g, TPB, 0, stream>>>(t_wq, t_wqT, DIM, DIM);
    wtrans_kernel<<<g, TPB, 0, stream>>>(t_wk, t_wkT, DIM, DIM);
    wtrans_kernel<<<g, TPB, 0, stream>>>(t_wv, t_wvT, DIM, DIM);
    wtrans_kernel<<<g, TPB, 0, stream>>>(t_wo, t_woT, DIM, DIM);
    int nm = DIM * MLPD, gm = (nm + TPB - 1) / TPB;
    wtrans_kernel<<<gm, TPB, 0, stream>>>(w1, w1T, DIM, MLPD);
    wtrans_kernel<<<gm, TPB, 0, stream>>>(w2, w2T, MLPD, DIM);
  }

  const int M_SP = BB * TT * HWN;   // 32768
  const int M_T  = BB * HWN * TQ;   // 36864

  auto gemm_blocks = [](int M, int N) { return ((M / 16) * (N / 16) + 7) / 8; };

  // ---------------- Phase A: spatial windowed attention ----------------
  ln_kernel<<<M_SP, TPB, 0, stream>>>(query, sln_g, sln_b, LN, 1);

  gemm_wmma<<<gemm_blocks(M_SP, DIM), TPB, 0, stream>>>(LN, s_wqT, s_bq, Qb, nullptr, nullptr, M_SP, DIM, DIM, 0, 0);
  gemm_wmma<<<gemm_blocks(M_SP, DIM), TPB, 0, stream>>>(LN, s_wkT, s_bk, Kb, nullptr, nullptr, M_SP, DIM, DIM, 0, 0);
  gemm_wmma<<<gemm_blocks(M_SP, DIM), TPB, 0, stream>>>(LN, s_wvT, s_bv, Vb, nullptr, nullptr, M_SP, DIM, DIM, 0, 0);

  spatial_attn<<<(M_SP * NH) / 8, TPB, 0, stream>>>(Qb, Kb, Vb, Ob);

  // O-proj + residual -> X (frames 0..7, remapped into T=9 layout)
  gemm_wmma<<<gemm_blocks(M_SP, DIM), TPB, 0, stream>>>(Ob, s_woT, s_bo, nullptr, X, query, M_SP, DIM, DIM, 1, 0);
  copy_last_frame<<<(BB * HWN * DIM + TPB - 1) / TPB, TPB, 0, stream>>>(query, X);

  // ---------------- Phase B: temporal RoPE attention ----------------
  ln_kernel<<<M_T, TPB, 0, stream>>>(X, tln_g, tln_b, LN, 2);

  gemm_wmma<<<gemm_blocks(M_T, DIM), TPB, 0, stream>>>(LN, t_wqT, t_bq, Qb, nullptr, nullptr, M_T, DIM, DIM, 0, 0);
  gemm_wmma<<<gemm_blocks(M_T, DIM), TPB, 0, stream>>>(LN, t_wkT, t_bk, Kb, nullptr, nullptr, M_T, DIM, DIM, 0, 0);
  gemm_wmma<<<gemm_blocks(M_T, DIM), TPB, 0, stream>>>(LN, t_wvT, t_bv, Vb, nullptr, nullptr, M_T, DIM, DIM, 0, 0);

  rope_kernel<<<(M_T * NH * 16 + TPB - 1) / TPB, TPB, 0, stream>>>(Qb, Kb);

  temporal_attn<<<(BB * HWN * NH) / 8, TPB, 0, stream>>>(Qb, Kb, Vb, tmask, Ob);

  // O-proj + residual -> out (acts as X2), remapped back to (b,t,p) layout
  gemm_wmma<<<gemm_blocks(M_T, DIM), TPB, 0, stream>>>(Ob, t_woT, t_bo, nullptr, out, X, M_T, DIM, DIM, 2, 0);

  // CLS frame: average over spatial, broadcast
  cls_avg<<<BB, TPB, 0, stream>>>(out, AVG);
  cls_bcast<<<(BB * HWN * DIM + TPB - 1) / TPB, TPB, 0, stream>>>(out, AVG);

  // ---------------- Phase C: MLP ----------------
  ln_kernel<<<M_T, TPB, 0, stream>>>(out, mln_g, mln_b, LN, 0);

  // GEMM1 + exact GELU -> hidden (f16), N=1024
  gemm_wmma<<<gemm_blocks(M_T, MLPD), TPB, 0, stream>>>(LN, w1T, b1, Hb, nullptr, nullptr, M_T, MLPD, DIM, 0, 1);

  // GEMM2 + bias + residual (out += ...), K=1024
  gemm_wmma<<<gemm_blocks(M_T, DIM), TPB, 0, stream>>>(Hb, w2T, b2, nullptr, out, out, M_T, DIM, MLPD, 0, 0);
}
